// FPE_16_16174846836759
// MI455X (gfx1250) — matrix operand order repaired and checked by reading, not compiled, not run on hardware
//
#include <hip/hip_runtime.h>
#include <hip/hip_bf16.h>
#include <math.h>

// Harness contract: kernel_launch only, all launches on `stream`, scratch in d_ws.

#define B_TOTAL   65536
#define NW        3      // windows
#define NF        48     // feats
#define EDIM      19     // MHA embed = 3 + 16
#define KPAD      64     // padded K for WMMA (57 -> 64)
#define NOUT      160    // encoder out = 16 hosts * 10 latent

typedef __attribute__((ext_vector_type(16))) _Float16 v16h;
typedef __attribute__((ext_vector_type(8)))  float    v8f;

__device__ __forceinline__ float sigmoidf_(float x) { return 1.0f / (1.0f + __expf(-x)); }

// ---------------------------------------------------------------------------
// Kernel 0: repack enc_W (160 x 57, f32 row-major) into f16 [n][k] with k
// padded to 64 (zeros) — this is the WMMA B operand source.
// ---------------------------------------------------------------------------
__global__ void prep_encw(const float* __restrict__ encW, _Float16* __restrict__ encBh) {
  int i = blockIdx.x * blockDim.x + threadIdx.x;
  if (i < NOUT * KPAD) {
    int n = i >> 6, k = i & 63;
    encBh[i] = (k < 57) ? (_Float16)encW[n * 57 + k] : (_Float16)0.0f;
  }
}

// ---------------------------------------------------------------------------
// Kernel 1: fused GRU + GAT + MHA, one thread per sample. Writes flat[b][0..56]
// as f16 padded to 64 columns (WMMA A operand).
// ---------------------------------------------------------------------------
__global__ __launch_bounds__(128) void fused_front(
    const float* __restrict__ t, const float* __restrict__ h0p,
    const float* __restrict__ gru_Wih, const float* __restrict__ gru_Whh,
    const float* __restrict__ gru_bih, const float* __restrict__ gru_bhh,
    const float* __restrict__ gat_W,  const float* __restrict__ gat_al,
    const float* __restrict__ gat_ar,
    const float* __restrict__ mha_Win, const float* __restrict__ mha_bin,
    const float* __restrict__ mha_Wout, const float* __restrict__ mha_bout,
    _Float16* __restrict__ flatOut)
{
  __shared__ float sWih[9 * NF];
  __shared__ float sWhh[9 * 3];
  __shared__ float sbih[9], sbhh[9];
  __shared__ float sGatW[3 * 16];
  __shared__ float sGWal[3], sGWar[3];
  __shared__ float sWin[57 * EDIM], sbin[57];
  __shared__ float sWout[EDIM * EDIM], sbout[EDIM];

  const int tid = threadIdx.x;
  for (int i = tid; i < 9 * NF; i += blockDim.x) sWih[i] = gru_Wih[i];
  for (int i = tid; i < 27;     i += blockDim.x) sWhh[i] = gru_Whh[i];
  if (tid < 9) { sbih[tid] = gru_bih[tid]; sbhh[tid] = gru_bhh[tid]; }
  for (int i = tid; i < 48; i += blockDim.x) sGatW[i] = gat_W[i];
  if (tid < 3) {  // fold gat_W @ al / ar into 3-vectors
    float a = 0.f, r = 0.f;
    for (int f = 0; f < 16; ++f) { a += gat_W[tid*16+f] * gat_al[f]; r += gat_W[tid*16+f] * gat_ar[f]; }
    sGWal[tid] = a; sGWar[tid] = r;
  }
  for (int i = tid; i < 57 * EDIM;   i += blockDim.x) sWin[i]  = mha_Win[i];
  if (tid < 57) sbin[tid] = mha_bin[tid];
  for (int i = tid; i < EDIM * EDIM; i += blockDim.x) sWout[i] = mha_Wout[i];
  if (tid < EDIM) sbout[tid] = mha_bout[tid];
  __syncthreads();

  const int b = blockIdx.x * blockDim.x + tid;
  const float* trow = t + (size_t)b * (NW * NF);

  float h[3];
  h[0] = h0p[b*3+0]; h[1] = h0p[b*3+1]; h[2] = h0p[b*3+2];

  float c[NW * EDIM];

  for (int w = 0; w < NW; ++w) {
    // ---- load x (48 floats, 16B aligned) ----
    float x[NF];
    #pragma unroll
    for (int i = 0; i < NF / 4; ++i) {
      float4 v4 = ((const float4*)(trow + w * NF))[i];
      x[4*i+0] = v4.x; x[4*i+1] = v4.y; x[4*i+2] = v4.z; x[4*i+3] = v4.w;
    }

    // ---- GRU step ----
    float gi[9], gh[9];
    #pragma unroll
    for (int j = 0; j < 9; ++j) {
      float s = sbih[j];
      #pragma unroll
      for (int i = 0; i < NF; ++i) s += x[i] * sWih[j * NF + i];
      gi[j] = s;
      gh[j] = sbhh[j] + h[0]*sWhh[j*3+0] + h[1]*sWhh[j*3+1] + h[2]*sWhh[j*3+2];
    }
    float hn[3];
    #pragma unroll
    for (int k2 = 0; k2 < 3; ++k2) {
      float r = sigmoidf_(gi[k2]     + gh[k2]);
      float z = sigmoidf_(gi[3 + k2] + gh[3 + k2]);
      float n = tanhf(gi[6 + k2] + r * gh[6 + k2]);
      hn[k2] = (1.0f - z) * n + z * h[k2];
    }
    #pragma unroll
    for (int k2 = 0; k2 < 3; ++k2) { h[k2] = hn[k2]; c[w * EDIM + k2] = hn[k2]; }

    // ---- GAT (algebraically reduced) ----
    float el[16], er[16];
    #pragma unroll
    for (int u = 0; u < 16; ++u) {
      el[u] = x[u*3]*sGWal[0] + x[u*3+1]*sGWal[1] + x[u*3+2]*sGWal[2];
      er[u] = x[u*3]*sGWar[0] + x[u*3+1]*sGWar[1] + x[u*3+2]*sGWar[2];
    }
    float wacc[16];
    #pragma unroll
    for (int u = 0; u < 16; ++u) wacc[u] = 0.0f;
    #pragma unroll
    for (int v = 0; v < 16; ++v) {
      float ev[16], mx = -3.4e38f;
      #pragma unroll
      for (int u = 0; u < 16; ++u) {
        float e = er[v] + el[u];
        e = (e > 0.0f) ? e : 0.2f * e;     // leaky_relu 0.2
        ev[u] = e; mx = fmaxf(mx, e);
      }
      float s = 0.0f;
      #pragma unroll
      for (int u = 0; u < 16; ++u) { ev[u] = __expf(ev[u] - mx); s += ev[u]; }
      float inv = 1.0f / (16.0f * s);      // fold mean over v
      #pragma unroll
      for (int u = 0; u < 16; ++u) wacc[u] += ev[u] * inv;
    }
    float gc0 = 0.f, gc1 = 0.f, gc2 = 0.f;
    #pragma unroll
    for (int u = 0; u < 16; ++u) {
      gc0 += wacc[u] * x[u*3]; gc1 += wacc[u] * x[u*3+1]; gc2 += wacc[u] * x[u*3+2];
    }
    #pragma unroll
    for (int f = 0; f < 16; ++f)
      c[w * EDIM + 3 + f] = gc0 * sGatW[f] + gc1 * sGatW[16 + f] + gc2 * sGatW[32 + f];
  }

  // ---- MHA over 3 tokens ----
  float kk[NW * EDIM], vv[NW * EDIM];
  #pragma unroll
  for (int s = 0; s < NW; ++s) {
    #pragma unroll
    for (int e = 0; e < EDIM; ++e) {
      float sk = sbin[EDIM + e], sv = sbin[2 * EDIM + e];
      #pragma unroll
      for (int f = 0; f < EDIM; ++f) {
        sk += c[s * EDIM + f] * sWin[(EDIM + e) * EDIM + f];
        sv += c[s * EDIM + f] * sWin[(2 * EDIM + e) * EDIM + f];
      }
      kk[s * EDIM + e] = sk; vv[s * EDIM + e] = sv;
    }
  }
  const float scale = 0.2294157339f;   // 1/sqrt(19)
  _Float16* fr = flatOut + (size_t)b * KPAD;
  #pragma unroll
  for (int s = 0; s < NW; ++s) {
    float q[EDIM];
    #pragma unroll
    for (int e = 0; e < EDIM; ++e) {
      float sq = sbin[e];
      #pragma unroll
      for (int f = 0; f < EDIM; ++f) sq += c[s * EDIM + f] * sWin[e * EDIM + f];
      q[e] = sq;
    }
    float sc[3];
    #pragma unroll
    for (int tt = 0; tt < 3; ++tt) {
      float d = 0.0f;
      #pragma unroll
      for (int e = 0; e < EDIM; ++e) d += q[e] * kk[tt * EDIM + e];
      sc[tt] = d * scale;
    }
    float mx = fmaxf(sc[0], fmaxf(sc[1], sc[2]));
    float e0 = __expf(sc[0]-mx), e1 = __expf(sc[1]-mx), e2 = __expf(sc[2]-mx);
    float inv = 1.0f / (e0 + e1 + e2);
    float a0 = e0*inv, a1 = e1*inv, a2 = e2*inv;
    float at[EDIM];
    #pragma unroll
    for (int e = 0; e < EDIM; ++e)
      at[e] = a0 * vv[e] + a1 * vv[EDIM + e] + a2 * vv[2 * EDIM + e];
    #pragma unroll
    for (int e = 0; e < EDIM; ++e) {
      float o = sbout[e];
      #pragma unroll
      for (int f = 0; f < EDIM; ++f) o += at[f] * sWout[e * EDIM + f];
      fr[s * EDIM + e] = (_Float16)o;
    }
  }
  #pragma unroll
  for (int j = 57; j < KPAD; ++j) fr[j] = (_Float16)0.0f;
}

// ---------------------------------------------------------------------------
// Kernel 2: WMMA GEMM  latent = flat(B x 57) @ enc_W.T (+enc_b), fused heads.
// 128 threads = 4 waves; each wave owns one 16-row tile of the batch.
// ---------------------------------------------------------------------------
__global__ __launch_bounds__(128) void gemm_heads(
    const _Float16* __restrict__ fA, const _Float16* __restrict__ encBh,
    const float* __restrict__ enc_b, const float* __restrict__ anom_W,
    const float* __restrict__ anom_b, const float* __restrict__ proto_W,
    const float* __restrict__ proto_b, float* __restrict__ out)
{
  __shared__ _Float16 sB[NOUT * KPAD];     // 20 KB
  __shared__ float    sD[4][16][NOUT];     // 40 KB, one 16x160 tile per wave
  __shared__ float    sEb[NOUT], sAW[20], sABv[2], sPW[100], sPB[10];

  const int tid = threadIdx.x;
  for (int i = tid; i < NOUT * KPAD; i += 128) sB[i] = encBh[i];
  for (int i = tid; i < NOUT;        i += 128) sEb[i] = enc_b[i];
  if (tid < 20)  sAW[tid]  = anom_W[tid];
  if (tid < 2)   sABv[tid] = anom_b[tid];
  if (tid < 100) sPW[tid]  = proto_W[tid];
  if (tid < 10)  sPB[tid]  = proto_b[tid];
  __syncthreads();

  const int wave = tid >> 5, lane = tid & 31;
  const int tile = blockIdx.x * 4 + wave;
  const int b0   = tile * 16;
  const int lhi  = lane >> 4;        // half-wave select
  const int llo  = lane & 15;
  const int kbA  = lhi * 8;          // A-frag K base per §7.12.2

  v8f zero8 = {0.f, 0.f, 0.f, 0.f, 0.f, 0.f, 0.f, 0.f};
  v8f acc[10];
  #pragma unroll
  for (int nt = 0; nt < 10; ++nt) acc[nt] = zero8;

  const _Float16* arow = fA + (size_t)(b0 + llo) * KPAD;   // A lane = row M

  #pragma unroll
  for (int k0 = 0; k0 < KPAD; k0 += 32) {
    v16h a;
    #pragma unroll
    for (int i = 0; i < 8; ++i) {
      a[i]     = arow[k0 + kbA + i];        // K = k0+kb .. k0+kb+7
      a[8 + i] = arow[k0 + 16 + kbA + i];   // K = k0+16+kb ..
    }
    const int kbB = k0 + lhi * 8;           // B: same K order as A (kb..kb+7, 16+kb..16+kb+7)
    #pragma unroll
    for (int nt = 0; nt < 10; ++nt) {
      const _Float16* bp = &sB[(nt * 16 + llo) * KPAD + kbB];  // lane = col N
      v16h bf;
      #pragma unroll
      for (int i = 0; i < 16; ++i) bf[i] = bp[(i & 7) + ((i >> 3) << 4)];
      acc[nt] = __builtin_amdgcn_wmma_f32_16x16x32_f16(
          false, a, false, bf, (short)0, acc[nt], false, false);
    }
  }

  // D layout: lane holds (m = i + 8*lhi, n = nt*16 + llo)
  #pragma unroll
  for (int nt = 0; nt < 10; ++nt)
    #pragma unroll
    for (int i = 0; i < 8; ++i)
      sD[wave][i + 8 * lhi][nt * 16 + llo] = acc[nt][i];
  __syncthreads();

  // Heads: lane -> (row = llo, hosts lhi*8 .. lhi*8+7)
  const int row = llo;
  const int hB  = lhi * 8;
  const int b   = b0 + row;
  const size_t protoOff = (size_t)B_TOTAL * 32;

  for (int h = hB; h < hB + 8; ++h) {
    float lat[10];
    #pragma unroll
    for (int j = 0; j < 10; ++j) lat[j] = sD[wave][row][h * 10 + j] + sEb[h * 10 + j];

    float a0 = sABv[0], a1 = sABv[1];
    #pragma unroll
    for (int j = 0; j < 10; ++j) { a0 += sAW[j] * lat[j]; a1 += sAW[10 + j] * lat[j]; }
    float mx = fmaxf(a0, a1);
    float e0 = __expf(a0 - mx), e1 = __expf(a1 - mx);
    float inv = 1.0f / (e0 + e1);
    out[(size_t)(b * 16 + h) * 2 + 0] = e0 * inv;
    out[(size_t)(b * 16 + h) * 2 + 1] = e1 * inv;

    #pragma unroll
    for (int p = 0; p < 10; ++p) {
      float s = sPB[p];
      #pragma unroll
      for (int j = 0; j < 10; ++j) s += sPW[p * 10 + j] * lat[j];
      out[protoOff + (size_t)(b * 16 + h) * 10 + p] = sigmoidf_(s);
    }
  }
}

// ---------------------------------------------------------------------------
extern "C" void kernel_launch(void* const* d_in, const int* in_sizes, int n_in,
                              void* d_out, int out_size, void* d_ws, size_t ws_size,
                              hipStream_t stream) {
  const float* t        = (const float*)d_in[0];
  // d_in[1] = s (unused scalar)
  const float* h0       = (const float*)d_in[2];
  const float* gru_Wih  = (const float*)d_in[3];
  const float* gru_Whh  = (const float*)d_in[4];
  const float* gru_bih  = (const float*)d_in[5];
  const float* gru_bhh  = (const float*)d_in[6];
  const float* gat_W    = (const float*)d_in[7];
  const float* gat_al   = (const float*)d_in[8];
  const float* gat_ar   = (const float*)d_in[9];
  const float* mha_Win  = (const float*)d_in[10];
  const float* mha_bin  = (const float*)d_in[11];
  const float* mha_Wout = (const float*)d_in[12];
  const float* mha_bout = (const float*)d_in[13];
  const float* enc_W    = (const float*)d_in[14];
  const float* enc_b    = (const float*)d_in[15];
  const float* anom_W   = (const float*)d_in[16];
  const float* anom_b   = (const float*)d_in[17];
  const float* proto_W  = (const float*)d_in[18];
  const float* proto_b  = (const float*)d_in[19];

  _Float16* encBh = (_Float16*)d_ws;                          // 160*64*2 = 20 KB
  _Float16* flatH = (_Float16*)((char*)d_ws + 32768);         // 65536*64*2 = 8 MB

  prep_encw<<<(NOUT * KPAD + 255) / 256, 256, 0, stream>>>(enc_W, encBh);

  fused_front<<<B_TOTAL / 128, 128, 0, stream>>>(
      t, h0, gru_Wih, gru_Whh, gru_bih, gru_bhh,
      gat_W, gat_al, gat_ar, mha_Win, mha_bin, mha_Wout, mha_bout, flatH);

  gemm_heads<<<B_TOTAL / 16 / 4, 128, 0, stream>>>(
      flatH, encBh, enc_b, anom_W, anom_b, proto_W, proto_b, (float*)d_out);
}
